// StructuralStreamGAT_51445118271854
// MI455X (gfx1250) — matrix operand order repaired and checked by reading, not compiled, not run on hardware
//
#include <hip/hip_runtime.h>
#include <hip/hip_bf16.h>

// ---------------------------------------------------------------------------
// CDNA5 (gfx1250) GAT forward.
//   GEMMs via v_wmma_f32_16x16x32_bf16 (wave32), B tiles streamed to LDS with
//   GLOBAL_LOAD_ASYNC_TO_LDS_B128 (ASYNCcnt), scatter phase via native f32
//   global atomics, LayerNorm via wave32 shuffle reductions.
// ---------------------------------------------------------------------------

typedef __attribute__((ext_vector_type(16))) __bf16 v16bf;
typedef __attribute__((ext_vector_type(8)))  float  v8f;
typedef __attribute__((ext_vector_type(4)))  int    v4i;

// pointee-typed address-space typedefs (attribute binds to the pointee type)
typedef __attribute__((address_space(1))) v4i gv4i;   // global int4
typedef __attribute__((address_space(3))) v4i lv4i;   // LDS int4

__device__ __forceinline__ unsigned short f2bf(float f) {
  unsigned u = __float_as_uint(f);
  u += 0x7FFFu + ((u >> 16) & 1u);      // round-to-nearest-even
  return (unsigned short)(u >> 16);
}

__device__ __forceinline__ void async_b128_to_lds(const void* g, void* l) {
#if __has_builtin(__builtin_amdgcn_global_load_async_to_lds_b128)
  __builtin_amdgcn_global_load_async_to_lds_b128((gv4i*)g, (lv4i*)l, 0, 0);
#else
  unsigned lds_off = (unsigned)(unsigned long long)l;
  asm volatile("global_load_async_to_lds_b128 %0, %1, off"
               :: "v"(lds_off), "v"(g) : "memory");
#endif
}

__device__ __forceinline__ void wait_async0() {
#if __has_builtin(__builtin_amdgcn_s_wait_asynccnt)
  __builtin_amdgcn_s_wait_asynccnt(0);
#else
  asm volatile("s_wait_asynccnt 0x0" ::: "memory");
#endif
}

// order-preserving float -> uint key for atomicMax-based segment max
__device__ __forceinline__ unsigned fkey(float f) {
  int i = __float_as_int(f);
  return (unsigned)(i >= 0 ? (i | 0x80000000) : ~i);
}
__device__ __forceinline__ float funkey(unsigned k) {
  int i = (k >> 31) ? (int)(k & 0x7FFFFFFFu) : ~(int)k;
  return __int_as_float(i);
}

// f32 -> bf16 conversion (for weight matrices, done once)
__global__ void convert_bf16_kernel(const float* __restrict__ src,
                                    unsigned short* __restrict__ dst, int n)
{
  const int i = blockIdx.x * blockDim.x + threadIdx.x;
  if (i < n) dst[i] = f2bf(src[i]);
}

// ---------------------------------------------------------------------------
// bf16 WMMA GEMM: C[M,Nc] = A[M,K](f32) @ B[K,Nc](bf16) (+ optional col bias)
// Block = 256 threads (8 waves), tile 128x64, K-step 32.
// Wave tile 32x32 -> 4x v_wmma_f32_16x16x32_bf16 per K-step.
// A tile: f32 global -> registers -> bf16 -> ds_store (handles M tail).
// B tile: bf16 global -> LDS via global_load_async_to_lds_b128 (ASYNCcnt).
// ---------------------------------------------------------------------------
__global__ __launch_bounds__(256)
void gemm_bf16_wmma(const float* __restrict__ A,
                    const unsigned short* __restrict__ B,
                    float* __restrict__ C, const float* __restrict__ bias,
                    int M, int K, int Nc)
{
  __shared__ unsigned short               lds_a[128][34]; // 17-dword rows (odd: no bank conflict)
  __shared__ __align__(16) unsigned short lds_b[32][72];  // 144B rows (16B aligned, 36 dwords)

  const int t    = threadIdx.x;
  const int wave = t >> 5;
  const int lane = t & 31;
  const int m0   = blockIdx.x * 128;
  const int n0   = blockIdx.y * 64;

  const int wm = (wave & 3) * 32;   // wave row offset inside block tile
  const int wn = (wave >> 2) * 32;  // wave col offset inside block tile

  v8f c00 = {}, c01 = {}, c10 = {}, c11 = {};

  // staging thread mapping
  const int ar = t >> 1;            // A: 128 rows, 2 threads/row
  const int ac = (t & 1) * 16;      // 16 floats each
  const int br = t >> 3;            // B: 32 rows, 8 threads/row
  const int bc = (t & 7) * 8;       // 8 bf16 (16 bytes) each -> one b128

  for (int k0 = 0; k0 < K; k0 += 32) {
    // ---- B tile (32x64 bf16): one async b128 per thread ----
    async_b128_to_lds(B + (size_t)(k0 + br) * Nc + (n0 + bc), &lds_b[br][bc]);

    // ---- A tile (128x32): load f32, convert, ds_store ----
    {
      const int gm = m0 + ar;
      if (gm < M) {
        const float* p = A + (size_t)gm * K + (k0 + ac);
        if (k0 + 32 < K) __builtin_prefetch(p + 32, 0, 3);
        #pragma unroll
        for (int j = 0; j < 16; ++j) lds_a[ar][ac + j] = f2bf(p[j]);
      } else {
        #pragma unroll
        for (int j = 0; j < 16; ++j) lds_a[ar][ac + j] = 0;
      }
    }
    wait_async0();
    __syncthreads();

    // ---- fragments (CDNA5 wave32 layouts) ----
    union { v16bf v; unsigned u[8]; } fa0, fa1, fb0, fb1;
    const int half = lane >> 4;
    const int mr0  = wm + (lane & 15);
    #pragma unroll
    for (int vg = 0; vg < 8; ++vg) {
      // 16-bit A 16x32: vg<4 -> K=2vg(+1)+8*half ; vg>=4 -> K=16+2(vg-4)(+1)+8*half
      const int kk = ((vg < 4) ? (2 * vg) : (16 + 2 * (vg - 4))) + half * 8;
      fa0.u[vg] = *(const unsigned*)&lds_a[mr0][kk];
      fa1.u[vg] = *(const unsigned*)&lds_a[mr0 + 16][kk];
    }
    #pragma unroll
    for (int j = 0; j < 8; ++j) {      // B 32x16: lane = N col, elems = K (same order as A)
      const int kk = ((j < 4) ? (2 * j) : (16 + 2 * (j - 4))) + half * 8;
      const int nb = wn + (lane & 15);
      fb0.u[j] = (unsigned)lds_b[kk][nb] | ((unsigned)lds_b[kk + 1][nb] << 16);
      fb1.u[j] = (unsigned)lds_b[kk][nb + 16] | ((unsigned)lds_b[kk + 1][nb + 16] << 16);
    }

    c00 = __builtin_amdgcn_wmma_f32_16x16x32_bf16(false, fa0.v, false, fb0.v, (short)0, c00, false, false);
    c01 = __builtin_amdgcn_wmma_f32_16x16x32_bf16(false, fa0.v, false, fb1.v, (short)0, c01, false, false);
    c10 = __builtin_amdgcn_wmma_f32_16x16x32_bf16(false, fa1.v, false, fb0.v, (short)0, c10, false, false);
    c11 = __builtin_amdgcn_wmma_f32_16x16x32_bf16(false, fa1.v, false, fb1.v, (short)0, c11, false, false);
    __syncthreads();
  }

  // ---- store: vgpr r -> row = r + 8*(lane>>4), col = lane&15 ----
  const int half = lane >> 4;
  const int colg = n0 + wn + (lane & 15);
  const float b0 = bias ? bias[colg]      : 0.f;
  const float b1 = bias ? bias[colg + 16] : 0.f;
  #pragma unroll
  for (int r = 0; r < 8; ++r) {
    const int row0 = m0 + wm + r + 8 * half;
    const int row1 = row0 + 16;
    if (row0 < M) {
      C[(size_t)row0 * Nc + colg]      = c00[r] + b0;
      C[(size_t)row0 * Nc + colg + 16] = c01[r] + b1;
    }
    if (row1 < M) {
      C[(size_t)row1 * Nc + colg]      = c10[r] + b0;
      C[(size_t)row1 * Nc + colg + 16] = c11[r] + b1;
    }
  }
}

// ---------------------------------------------------------------------------
// per-(node, head) attention scores
// ---------------------------------------------------------------------------
__global__ void att_scores_kernel(const float* __restrict__ h,
                                  const float* __restrict__ att_s,
                                  const float* __restrict__ att_d,
                                  float* __restrict__ a_s, float* __restrict__ a_d,
                                  int N, int H)
{
  const int i = blockIdx.x * blockDim.x + threadIdx.x;
  if (i >= N * H) return;
  const int n = i / H, hd = i % H;
  const float* hp = h + (size_t)n * H * 64 + hd * 64;
  const float* as = att_s + hd * 64;
  const float* ad = att_d + hd * 64;
  float ss = 0.f, sd = 0.f;
  #pragma unroll 8
  for (int c = 0; c < 64; ++c) { float v = hp[c]; ss += v * as[c]; sd += v * ad[c]; }
  a_s[i] = ss; a_d[i] = sd;
}

__device__ __forceinline__ int edge_src(const int* ei, int E, int e) {
  return e < E ? ei[e] : e - E;           // self-loops appended
}
__device__ __forceinline__ int edge_dst(const int* ei, int E, int e) {
  return e < E ? ei[E + e] : e - E;
}

// pass 1: leaky-relu logits + segment max via ordered-key atomicMax
__global__ void edge_logits_kernel(const int* __restrict__ ei, int E, int Etot, int H,
                                   const float* __restrict__ a_s,
                                   const float* __restrict__ a_d,
                                   float* __restrict__ logits,
                                   unsigned* __restrict__ amax_key)
{
  const int i = blockIdx.x * blockDim.x + threadIdx.x;
  if (i >= Etot * H) return;
  const int e = i / H, hd = i % H;
  const int s = edge_src(ei, E, e);
  const int d = edge_dst(ei, E, e);
  float a = a_s[s * H + hd] + a_d[d * H + hd];
  a = (a >= 0.f) ? a : 0.2f * a;          // LeakyReLU(0.2)
  logits[i] = a;
  atomicMax(&amax_key[d * H + hd], fkey(a));
}

// pass 2: exp(logit - max) + segment sum
__global__ void edge_exp_kernel(const int* __restrict__ ei, int E, int Etot, int H,
                                float* __restrict__ logits,
                                const unsigned* __restrict__ amax_key,
                                float* __restrict__ denom)
{
  const int i = blockIdx.x * blockDim.x + threadIdx.x;
  if (i >= Etot * H) return;
  const int e = i / H, hd = i % H;
  const int d = edge_dst(ei, E, e);
  const float mx = funkey(amax_key[d * H + hd]);
  const float ex = __expf(logits[i] - mx);
  logits[i] = ex;
  atomicAdd(&denom[d * H + hd], ex);
}

// pass 3: weighted scatter of messages (32 lanes per edge)
__global__ __launch_bounds__(256)
void edge_scatter_kernel(const int* __restrict__ ei, int E, int Etot, int H,
                         const float* __restrict__ h, const float* __restrict__ ex,
                         const float* __restrict__ denom, float* __restrict__ out)
{
  const int e    = blockIdx.x * 8 + (threadIdx.x >> 5);
  const int lane = threadIdx.x & 31;
  if (e >= Etot) return;
  const int s = edge_src(ei, E, e);
  const int d = edge_dst(ei, E, e);
  float coef[4];
  for (int hd = 0; hd < H; ++hd)
    coef[hd] = ex[e * H + hd] / (denom[d * H + hd] + 1e-16f);
  const int HC = H * 64;
  for (int f = lane; f < HC; f += 32) {
    const int hd = f >> 6;
    atomicAdd(&out[(size_t)d * HC + f], h[(size_t)s * HC + f] * coef[hd]);
  }
}

// bias + ELU in place
__global__ void bias_elu_kernel(float* __restrict__ x, const float* __restrict__ bias,
                                size_t total, int F)
{
  const size_t i = (size_t)blockIdx.x * blockDim.x + threadIdx.x;
  if (i >= total) return;
  const float v = x[i] + bias[i % F];
  x[i] = (v > 0.f) ? v : (__expf(v) - 1.f);
}

// LayerNorm(64) over (acc + bias), one wave32 per node, shuffle reductions
__global__ __launch_bounds__(256)
void layernorm_kernel(const float* __restrict__ acc, const float* __restrict__ bias,
                      const float* __restrict__ gamma, const float* __restrict__ beta,
                      float* __restrict__ out, int N)
{
  const int n    = blockIdx.x * 8 + (threadIdx.x >> 5);
  const int lane = threadIdx.x & 31;
  if (n >= N) return;
  const float v0 = acc[(size_t)n * 64 + lane]      + bias[lane];
  const float v1 = acc[(size_t)n * 64 + lane + 32] + bias[lane + 32];
  float s = v0 + v1;
  #pragma unroll
  for (int m = 16; m >= 1; m >>= 1) s += __shfl_xor(s, m, 32);
  const float mu = s * (1.f / 64.f);
  const float d0 = v0 - mu, d1 = v1 - mu;
  float vs = d0 * d0 + d1 * d1;
  #pragma unroll
  for (int m = 16; m >= 1; m >>= 1) vs += __shfl_xor(vs, m, 32);
  const float rstd = rsqrtf(vs * (1.f / 64.f) + 1e-5f);
  out[(size_t)n * 64 + lane]      = d0 * rstd * gamma[lane]      + beta[lane];
  out[(size_t)n * 64 + lane + 32] = d1 * rstd * gamma[lane + 32] + beta[lane + 32];
}

__global__ void zero_kernel(float* __restrict__ p, size_t n)
{
  size_t i = (size_t)blockIdx.x * blockDim.x + threadIdx.x;
  const size_t stride = (size_t)gridDim.x * blockDim.x;
  for (; i < n; i += stride) p[i] = 0.f;
}

// ---------------------------------------------------------------------------
extern "C" void kernel_launch(void* const* d_in, const int* in_sizes, int n_in,
                              void* d_out, int out_size, void* d_ws, size_t ws_size,
                              hipStream_t stream)
{
  const float* x        = (const float*)d_in[0];
  const int*   ei       = (const int*)  d_in[1];
  const float* W_in     = (const float*)d_in[2];
  const float* b_in     = (const float*)d_in[3];
  const float* W1       = (const float*)d_in[4];
  const float* att_src1 = (const float*)d_in[5];
  const float* att_dst1 = (const float*)d_in[6];
  const float* bias1    = (const float*)d_in[7];
  const float* W2       = (const float*)d_in[8];
  const float* att_src2 = (const float*)d_in[9];
  const float* att_dst2 = (const float*)d_in[10];
  const float* bias2    = (const float*)d_in[11];
  const float* gamma    = (const float*)d_in[12];
  const float* beta     = (const float*)d_in[13];

  const int N    = in_sizes[0] / 256;
  const int E    = in_sizes[1] / 2;
  const int Etot = E + N;

  // ---- workspace layout (with reuse) ----
  char* ws = (char*)d_ws;
  size_t off = 0;
  auto alloc = [&](size_t bytes) -> void* {
    void* p = ws + off;
    off = (off + bytes + 255) & ~(size_t)255;
    return p;
  };
  float*          h0     = (float*)         alloc((size_t)N * 64 * 4);   // reused as h2
  float*          h1     = (float*)         alloc((size_t)N * 256 * 4);  // reused as out2
  float*          out1   = (float*)         alloc((size_t)N * 256 * 4);
  float*          a_s1   = (float*)         alloc((size_t)N * 4 * 4);
  float*          a_d1   = (float*)         alloc((size_t)N * 4 * 4);
  unsigned*       amax1  = (unsigned*)      alloc((size_t)N * 4 * 4);
  float*          den1   = (float*)         alloc((size_t)N * 4 * 4);
  float*          a_s2   = (float*)         alloc((size_t)N * 4);
  float*          a_d2   = (float*)         alloc((size_t)N * 4);
  unsigned*       amax2  = (unsigned*)      alloc((size_t)N * 4);
  float*          den2   = (float*)         alloc((size_t)N * 4);
  float*          logit1 = (float*)         alloc((size_t)Etot * 4 * 4); // reused for layer 2
  unsigned short* Wb_in  = (unsigned short*)alloc(256 * 64 * 2);
  unsigned short* Wb1    = (unsigned short*)alloc(64 * 256 * 2);
  unsigned short* Wb2    = (unsigned short*)alloc(256 * 64 * 2);
  float*          h2     = h0;      // h0 dead after GEMM2
  float*          out2   = h1;      // h1 dead after layer-1 scatter
  float*          logit2 = logit1;  // logit1 dead after layer-1 scatter

  const dim3 blk(256);
  const int  gemmM = (N + 127) / 128;

  // 0) weights -> bf16 (tiny, L2-resident; enables raw async B-tile copies)
  convert_bf16_kernel<<<(256 * 64 + 255) / 256, blk, 0, stream>>>(W_in, Wb_in, 256 * 64);
  convert_bf16_kernel<<<(64 * 256 + 255) / 256, blk, 0, stream>>>(W1,   Wb1,   64 * 256);
  convert_bf16_kernel<<<(256 * 64 + 255) / 256, blk, 0, stream>>>(W2,   Wb2,   256 * 64);

  // 1) h0 = x @ W_in + b_in
  gemm_bf16_wmma<<<dim3(gemmM, 1), blk, 0, stream>>>(x, Wb_in, h0, b_in, N, 256, 64);

  // 2) h1 = h0 @ W1
  gemm_bf16_wmma<<<dim3(gemmM, 4), blk, 0, stream>>>(h0, Wb1, h1, nullptr, N, 64, 256);

  // 3) GAT layer 1 (H=4)
  att_scores_kernel<<<(N * 4 + 255) / 256, blk, 0, stream>>>(h1, att_src1, att_dst1, a_s1, a_d1, N, 4);
  zero_kernel<<<1024, blk, 0, stream>>>((float*)amax1, (size_t)N * 4);
  zero_kernel<<<1024, blk, 0, stream>>>(den1, (size_t)N * 4);
  zero_kernel<<<2048, blk, 0, stream>>>(out1, (size_t)N * 256);
  edge_logits_kernel<<<(Etot * 4 + 255) / 256, blk, 0, stream>>>(ei, E, Etot, 4, a_s1, a_d1, logit1, amax1);
  edge_exp_kernel<<<(Etot * 4 + 255) / 256, blk, 0, stream>>>(ei, E, Etot, 4, logit1, amax1, den1);
  edge_scatter_kernel<<<(Etot + 7) / 8, blk, 0, stream>>>(ei, E, Etot, 4, h1, logit1, den1, out1);
  bias_elu_kernel<<<((size_t)N * 256 + 255) / 256, blk, 0, stream>>>(out1, bias1, (size_t)N * 256, 256);

  // 4) h2 = out1 @ W2
  gemm_bf16_wmma<<<dim3(gemmM, 1), blk, 0, stream>>>(out1, Wb2, h2, nullptr, N, 256, 64);

  // 5) GAT layer 2 (H=1)
  att_scores_kernel<<<(N + 255) / 256, blk, 0, stream>>>(h2, att_src2, att_dst2, a_s2, a_d2, N, 1);
  zero_kernel<<<512, blk, 0, stream>>>((float*)amax2, (size_t)N);
  zero_kernel<<<512, blk, 0, stream>>>(den2, (size_t)N);
  zero_kernel<<<1024, blk, 0, stream>>>(out2, (size_t)N * 64);
  edge_logits_kernel<<<(Etot + 255) / 256, blk, 0, stream>>>(ei, E, Etot, 1, a_s2, a_d2, logit2, amax2);
  edge_exp_kernel<<<(Etot + 255) / 256, blk, 0, stream>>>(ei, E, Etot, 1, logit2, amax2, den2);
  edge_scatter_kernel<<<(Etot + 7) / 8, blk, 0, stream>>>(ei, E, Etot, 1, h2, logit2, den2, out2);

  // 6) LayerNorm(out2 + bias2)
  layernorm_kernel<<<(N + 7) / 8, blk, 0, stream>>>(out2, bias2, gamma, beta, (float*)d_out, N);
}
